// MultiHeadCrossAttention_74208444940648
// MI455X (gfx1250) — compile-verified, matrix operand order checked by reading, not run on hardware
//
#include <hip/hip_runtime.h>

#define EMBED 1024
#define HEADS 16
#define HDIM  64
#define BATCH 2
#define TSEQ  2048
#define MROWS (BATCH * TSEQ)   // 4096

typedef __attribute__((ext_vector_type(16))) __bf16 v16bf;
typedef __attribute__((ext_vector_type(8)))  float  v8f;

struct alignas(16) U128 { unsigned int w[4]; };
struct alignas(16) U256 { U128 lo, hi; };

__device__ __forceinline__ unsigned short f2bf(float f) {
  unsigned int u = __builtin_bit_cast(unsigned int, f);
  u += 0x7FFFu + ((u >> 16) & 1u);                 // round-to-nearest-even
  return (unsigned short)(u >> 16);
}

// Load one 16-element bf16 WMMA A/B fragment for this lane:
// 8 contiguous bf16 at p (K-chunk base) and 8 contiguous bf16 at p+16.
// Caller supplies p = row_base + kb where kb = (lane>=16 ? 8 : 0).
__device__ __forceinline__ v16bf load_frag(const unsigned short* p) {
  U256 t;
  t.lo = *reinterpret_cast<const U128*>(p);
  t.hi = *reinterpret_cast<const U128*>(p + 16);
  return __builtin_bit_cast(v16bf, t);
}

// ---------------------------------------------------------------------------
// Elementwise fp32 -> bf16
__global__ __launch_bounds__(256) void f32_to_bf16_kernel(
    const float* __restrict__ in, unsigned short* __restrict__ out, int n) {
  int i = blockIdx.x * 256 + threadIdx.x;
  if (i < n) out[i] = f2bf(in[i]);
}

// Transpose 1024x1024 fp32 weight -> bf16 [N,K] (row n = column n of W)
__global__ __launch_bounds__(256) void transpose_bf16_kernel(
    const float* __restrict__ in, unsigned short* __restrict__ out) {
  int i = blockIdx.x * 256 + threadIdx.x;   // 0 .. 1M-1
  int n = i >> 10, k = i & (EMBED - 1);
  out[i] = f2bf(in[k * EMBED + n]);
}

// ---------------------------------------------------------------------------
// bf16 WMMA GEMM:  C[M=4096, N=1024] = A[M,K=1024] * W,   Bt = W^T (bf16)
// Block tile 128x128 (8 waves), wave tile 32x64 = 2x4 WMMA C-fragments.
// mode 0: store bf16 into [B,H,T,64]      (Q / K head split)
// mode 1: store bf16 into [B,H,64,T]      (V transposed)
// mode 2: store f32  C + bias[N] + resid[M,N] into out
__global__ __launch_bounds__(256) void gemm_bf16_wmma(
    const unsigned short* __restrict__ A,
    const unsigned short* __restrict__ Bt,
    void* __restrict__ out,
    const float* __restrict__ bias,
    const float* __restrict__ resid,
    const int mode)
{
  const int lane = threadIdx.x & 31;
  const int wv   = threadIdx.x >> 5;    // wave 0..7
  const int lm   = lane & 15;
  const int hv   = lane >> 4;           // half-wave select

  const int m0 = blockIdx.x * 128 + (wv & 3) * 32;
  const int n0 = blockIdx.y * 128 + (wv >> 2) * 64;

  v8f c[2][4] = {};

  const unsigned short* a0p = A  + (m0 + lm) * EMBED + hv * 8;
  const unsigned short* a1p = a0p + 16 * EMBED;
  const unsigned short* b0p = Bt + (n0 + lm) * EMBED + hv * 8;
  const unsigned short* b1p = b0p + 16 * EMBED;
  const unsigned short* b2p = b0p + 32 * EMBED;
  const unsigned short* b3p = b0p + 48 * EMBED;

  for (int kt = 0; kt < EMBED; kt += 32) {
    v16bf a0 = load_frag(a0p + kt);
    v16bf a1 = load_frag(a1p + kt);
    v16bf b0 = load_frag(b0p + kt);
    v16bf b1 = load_frag(b1p + kt);
    v16bf b2 = load_frag(b2p + kt);
    v16bf b3 = load_frag(b3p + kt);
    c[0][0] = __builtin_amdgcn_wmma_f32_16x16x32_bf16(false, a0, false, b0, (short)0, c[0][0], false, false);
    c[0][1] = __builtin_amdgcn_wmma_f32_16x16x32_bf16(false, a0, false, b1, (short)0, c[0][1], false, false);
    c[0][2] = __builtin_amdgcn_wmma_f32_16x16x32_bf16(false, a0, false, b2, (short)0, c[0][2], false, false);
    c[0][3] = __builtin_amdgcn_wmma_f32_16x16x32_bf16(false, a0, false, b3, (short)0, c[0][3], false, false);
    c[1][0] = __builtin_amdgcn_wmma_f32_16x16x32_bf16(false, a1, false, b0, (short)0, c[1][0], false, false);
    c[1][1] = __builtin_amdgcn_wmma_f32_16x16x32_bf16(false, a1, false, b1, (short)0, c[1][1], false, false);
    c[1][2] = __builtin_amdgcn_wmma_f32_16x16x32_bf16(false, a1, false, b2, (short)0, c[1][2], false, false);
    c[1][3] = __builtin_amdgcn_wmma_f32_16x16x32_bf16(false, a1, false, b3, (short)0, c[1][3], false, false);
  }

#pragma unroll
  for (int ti = 0; ti < 2; ++ti)
#pragma unroll
    for (int tj = 0; tj < 4; ++tj)
#pragma unroll
      for (int r = 0; r < 8; ++r) {
        const int M = m0 + ti * 16 + r + hv * 8;
        const int N = n0 + tj * 16 + lm;
        const float acc = c[ti][tj][r];
        if (mode == 2) {
          float* o = reinterpret_cast<float*>(out);
          o[M * EMBED + N] = acc + bias[N] + resid[M * EMBED + N];
        } else {
          const int bb = M >> 11;            // batch (2048 rows each)
          const int t  = M & (TSEQ - 1);
          const int h  = N >> 6;             // head
          const int d  = N & (HDIM - 1);
          unsigned short* o = reinterpret_cast<unsigned short*>(out);
          const int idx = (mode == 0)
              ? (((bb * HEADS + h) * TSEQ + t) * HDIM + d)   // [B,H,T,64]
              : (((bb * HEADS + h) * HDIM + d) * TSEQ + t);  // [B,H,64,T]
          o[idx] = f2bf(acc);
        }
      }
}

// ---------------------------------------------------------------------------
// Flash attention: all 8 waves of a block share one (b,h); each wave owns a
// 16-query tile. K/V chunks are staged block-wide into LDS with
// global_load_async_to_lds_b128 (ASYNCcnt), then consumed as WMMA fragments
// via ds_load_b128. Online softmax over 32-key chunks.
__global__ __launch_bounds__(256) void attn_wmma_kernel(
    const unsigned short* __restrict__ Q,    // [B,H,TQ,64] bf16
    const unsigned short* __restrict__ K,    // [B,H,TK,64] bf16
    const unsigned short* __restrict__ Vt,   // [B,H,64,TK] bf16
    const float* __restrict__ pad,           // [B,TK] fp32
    unsigned short* __restrict__ O)          // [B*TQ, 1024] bf16
{
  __shared__ unsigned short sK[32 * HDIM];   // 4KB  [j-local][d]
  __shared__ unsigned short sV[HDIM * 32];   // 4KB  [d][k-local]
  __shared__ float          sS[8][512];      // per-wave 16x32 scores
  __shared__ unsigned short sP[8][512];      // per-wave 16x32 probs (bf16)
  __shared__ float          sA[8][16];       // per-wave per-row alpha / inv-l

  const int tid  = threadIdx.x;
  const int lane = tid & 31;
  const int wv   = tid >> 5;
  const int lm   = lane & 15;
  const int hv   = lane >> 4;

  // 16 blocks per (b,h); bh derived from blockIdx only -> uniform (SGPR).
  const int bh = blockIdx.x >> 4;
  const int q0 = (((blockIdx.x & 15) << 3) | wv) << 4;
  const int b  = bh / HEADS;
  const int h  = bh % HEADS;

  const unsigned short* Qh = Q + bh * TSEQ * HDIM;
  const unsigned long long KhA = (unsigned long long)(uintptr_t)(K  + bh * TSEQ * HDIM);
  const unsigned long long VtA = (unsigned long long)(uintptr_t)(Vt + bh * HDIM * TSEQ);
  const float* padb = pad + b * TSEQ;

  // Per-thread staging assignment: one b128 of K and one b128 of V per chunk.
  const int kRow = tid >> 3;           // 0..31 (key row in chunk)
  const int kCol = (tid & 7) * 8;      // 0..56 (d)
  const int vRow = tid >> 2;           // 0..63 (d)
  const int vCol = (tid & 3) * 8;      // 0..24 (k-local)
  // Flat-LDS aperture keeps the LDS byte offset in addr[31:0].
  const unsigned kLds = (unsigned)(uintptr_t)(sK + kRow * HDIM + kCol);
  const unsigned vLds = (unsigned)(uintptr_t)(sV + vRow * 32 + vCol);

  // Q A-fragments for d-chunks 0..31 and 32..63 (held for whole loop)
  const unsigned short* qrow = Qh + (q0 + lm) * HDIM + hv * 8;
  const v16bf qf0 = load_frag(qrow);
  const v16bf qf1 = load_frag(qrow + 32);

  v8f o[4] = {};                 // output d-tiles 0..3 (16 cols each)
  float mrow = -1e30f, lrow = 0.0f;

  float*          mySS = sS[wv];
  unsigned short* mySP = sP[wv];
  float*          myA  = sA[wv];

  for (int j0 = 0; j0 < TSEQ; j0 += 32) {
    __syncthreads();   // previous chunk's readers are done with sK/sV

    // ---- async stage K (32x64) and V^T (64x32) chunks into LDS ----
    const unsigned kOff = (unsigned)(((j0 + kRow) * HDIM + kCol) * 2);
    const unsigned vOff = (unsigned)((vRow * TSEQ + j0 + vCol) * 2);
    asm volatile("global_load_async_to_lds_b128 %0, %1, %2"
                 :: "v"(kLds), "v"(kOff), "s"(KhA) : "memory");
    asm volatile("global_load_async_to_lds_b128 %0, %1, %2"
                 :: "v"(vLds), "v"(vOff), "s"(VtA) : "memory");
    asm volatile("s_wait_asynccnt 0x0" ::: "memory");
    __syncthreads();   // staged data visible to all waves

    // ---- S = Q * K^T : B-fragment = A-layout applied to K rows (from LDS) ----
    const unsigned short* k0r = sK + lm * HDIM + hv * 8;
    const unsigned short* k1r = k0r + 16 * HDIM;
    v16bf kf00 = load_frag(k0r);        // keys j0..+15,  d 0..31
    v16bf kf01 = load_frag(k0r + 32);   // keys j0..+15,  d 32..63
    v16bf kf10 = load_frag(k1r);        // keys j0+16..31, d 0..31
    v16bf kf11 = load_frag(k1r + 32);

    v8f s0 = {}, s1 = {};
    s0 = __builtin_amdgcn_wmma_f32_16x16x32_bf16(false, qf0, false, kf00, (short)0, s0, false, false);
    s0 = __builtin_amdgcn_wmma_f32_16x16x32_bf16(false, qf1, false, kf01, (short)0, s0, false, false);
    s1 = __builtin_amdgcn_wmma_f32_16x16x32_bf16(false, qf0, false, kf10, (short)0, s1, false, false);
    s1 = __builtin_amdgcn_wmma_f32_16x16x32_bf16(false, qf1, false, kf11, (short)0, s1, false, false);

    // ---- scale + pad mask, spill scores to LDS in row-major 16x32 ----
    const float msk0 = padb[j0 + lm];
    const float msk1 = padb[j0 + 16 + lm];
#pragma unroll
    for (int r = 0; r < 8; ++r) {
      const int row = r + hv * 8;
      mySS[row * 32 + lm]      = s0[r] * 0.125f * msk0 - (1.0f - msk0) * 1e10f;
      mySS[row * 32 + 16 + lm] = s1[r] * 0.125f * msk1 - (1.0f - msk1) * 1e10f;
    }
    __syncthreads();

    // ---- online softmax row pass (lanes 0..15, one row each) ----
    if (lane < 16) {
      const float* rp = mySS + lane * 32;
      float mx = mrow;
#pragma unroll 8
      for (int c2 = 0; c2 < 32; ++c2) mx = fmaxf(mx, rp[c2]);
      const float alpha = __expf(mrow - mx);
      float sum = 0.0f;
#pragma unroll 8
      for (int c2 = 0; c2 < 32; ++c2) {
        const float p = __expf(rp[c2] - mx);
        sum += p;
        mySP[lane * 32 + c2] = f2bf(p);
      }
      lrow = lrow * alpha + sum;
      mrow = mx;
      myA[lane] = alpha;
    }
    __syncthreads();

    // ---- rescale accumulators by per-row alpha ----
#pragma unroll
    for (int r = 0; r < 8; ++r) {
      const float al = myA[r + hv * 8];
      o[0][r] *= al; o[1][r] *= al; o[2][r] *= al; o[3][r] *= al;
    }

    // ---- O += P * V : P as A-fragment from LDS, V^T rows from LDS ----
    const v16bf pf = load_frag(mySP + lm * 32 + hv * 8);
#pragma unroll
    for (int t = 0; t < 4; ++t) {
      const v16bf vf = load_frag(sV + (t * 16 + lm) * 32 + hv * 8);
      o[t] = __builtin_amdgcn_wmma_f32_16x16x32_bf16(false, pf, false, vf, (short)0, o[t], false, false);
    }
  }

  // ---- finalize: O /= l, write bf16 into [B*TQ, 1024] at column h*64+d ----
  __syncthreads();
  if (lane < 16) myA[lane] = 1.0f / lrow;
  __syncthreads();
#pragma unroll
  for (int r = 0; r < 8; ++r) {
    const float inv = myA[r + hv * 8];
    const int row = b * TSEQ + q0 + r + hv * 8;
#pragma unroll
    for (int t = 0; t < 4; ++t)
      O[row * EMBED + h * HDIM + t * 16 + lm] = f2bf(o[t][r] * inv);
  }
}

// ---------------------------------------------------------------------------
// Row layernorm over 1024 columns (population variance, eps 1e-5)
__global__ __launch_bounds__(256) void layernorm_kernel(
    const float* __restrict__ x, const float* __restrict__ gamma,
    const float* __restrict__ beta, float* __restrict__ out)
{
  __shared__ float rs[256], rs2[256];
  const int row = blockIdx.x;
  const float* xr = x + row * EMBED;
  float s = 0.0f, s2 = 0.0f;
  for (int i = threadIdx.x; i < EMBED; i += 256) {
    const float v = xr[i];
    s += v; s2 += v * v;
  }
  rs[threadIdx.x] = s; rs2[threadIdx.x] = s2;
  __syncthreads();
  for (int off = 128; off > 0; off >>= 1) {
    if (threadIdx.x < off) {
      rs[threadIdx.x]  += rs[threadIdx.x + off];
      rs2[threadIdx.x] += rs2[threadIdx.x + off];
    }
    __syncthreads();
  }
  const float mean = rs[0] * (1.0f / EMBED);
  const float var  = rs2[0] * (1.0f / EMBED) - mean * mean;
  const float inv  = rsqrtf(var + 1e-5f);
  for (int i = threadIdx.x; i < EMBED; i += 256)
    out[row * EMBED + i] = (xr[i] - mean) * inv * gamma[i] + beta[i];
}

// ---------------------------------------------------------------------------
extern "C" void kernel_launch(void* const* d_in, const int* in_sizes, int n_in,
                              void* d_out, int out_size, void* d_ws, size_t ws_size,
                              hipStream_t stream) {
  const float* target  = (const float*)d_in[0];
  const float* context = (const float*)d_in[1];
  const float* pad     = (const float*)d_in[2];
  const float* wq      = (const float*)d_in[3];
  const float* wk      = (const float*)d_in[4];
  const float* wv      = (const float*)d_in[5];
  const float* wo      = (const float*)d_in[6];
  const float* bo      = (const float*)d_in[7];
  const float* gamma   = (const float*)d_in[8];
  const float* beta    = (const float*)d_in[9];

  char* ws = (char*)d_ws;
  const size_t ACT_E = (size_t)MROWS * EMBED;              // 4,194,304 elems
  const size_t ACT_B = ACT_E * sizeof(unsigned short);     // 8 MB
  const size_t W_B   = (size_t)EMBED * EMBED * sizeof(unsigned short); // 2 MB

  unsigned short* tgt_bf = (unsigned short*)(ws);
  unsigned short* ctx_bf = (unsigned short*)(ws + ACT_B);
  unsigned short* wqT    = (unsigned short*)(ws + 2 * ACT_B);
  unsigned short* wkT    = (unsigned short*)(ws + 2 * ACT_B + W_B);
  unsigned short* wvT    = (unsigned short*)(ws + 2 * ACT_B + 2 * W_B);
  unsigned short* woT    = (unsigned short*)(ws + 2 * ACT_B + 3 * W_B);
  unsigned short* Qb     = (unsigned short*)(ws + 2 * ACT_B + 4 * W_B);
  unsigned short* Kb     = (unsigned short*)(ws + 3 * ACT_B + 4 * W_B);
  unsigned short* Vtb    = (unsigned short*)(ws + 4 * ACT_B + 4 * W_B);
  unsigned short* attnO  = (unsigned short*)(ws + 5 * ACT_B + 4 * W_B);
  float*          proj   = (float*)       (ws + 6 * ACT_B + 4 * W_B);
  // total workspace use: 6*8MB + 4*2MB + 16MB = 72 MB

  dim3 blk(256);
  f32_to_bf16_kernel<<<dim3((unsigned)(ACT_E / 256)), blk, 0, stream>>>(target,  tgt_bf, (int)ACT_E);
  f32_to_bf16_kernel<<<dim3((unsigned)(ACT_E / 256)), blk, 0, stream>>>(context, ctx_bf, (int)ACT_E);

  const unsigned WNB = (EMBED * EMBED) / 256;
  transpose_bf16_kernel<<<dim3(WNB), blk, 0, stream>>>(wq, wqT);
  transpose_bf16_kernel<<<dim3(WNB), blk, 0, stream>>>(wk, wkT);
  transpose_bf16_kernel<<<dim3(WNB), blk, 0, stream>>>(wv, wvT);
  transpose_bf16_kernel<<<dim3(WNB), blk, 0, stream>>>(wo, woT);

  dim3 ggrid(MROWS / 128, EMBED / 128);  // (32, 8)
  gemm_bf16_wmma<<<ggrid, blk, 0, stream>>>(tgt_bf, wqT, Qb,  nullptr, nullptr, 0);
  gemm_bf16_wmma<<<ggrid, blk, 0, stream>>>(ctx_bf, wkT, Kb,  nullptr, nullptr, 0);
  gemm_bf16_wmma<<<ggrid, blk, 0, stream>>>(ctx_bf, wvT, Vtb, nullptr, nullptr, 1);

  attn_wmma_kernel<<<dim3((BATCH * HEADS * (TSEQ / 16)) / 8), blk, 0, stream>>>(
      Qb, Kb, Vtb, pad, attnO);

  gemm_bf16_wmma<<<ggrid, blk, 0, stream>>>(attnO, woT, proj, bo, target, 2);

  layernorm_kernel<<<dim3(MROWS), blk, 0, stream>>>(proj, gamma, beta, (float*)d_out);

  (void)in_sizes; (void)n_in; (void)out_size; (void)ws_size;
}
